// Attention_83081847374268
// MI455X (gfx1250) — compile-verified, matrix operand order checked by reading, not run on hardware
//
#include <hip/hip_runtime.h>
#include <hip/hip_bf16.h>
#include <math.h>

// Problem constants (match reference)
#define BB   2
#define TT   2048
#define WID  2048
#define NH   8
#define NKV  4
#define HD   256
#define GRP  2
#define WIN  512
#define TP   2080            // time dim padded to multiple of 32 for v^T
#define BIG_NEG -2.3819763e38f

typedef __attribute__((ext_vector_type(16))) _Float16 v16h;
typedef __attribute__((ext_vector_type(8)))  _Float16 v8h;
typedef __attribute__((ext_vector_type(8)))  float    v8f;
typedef __attribute__((ext_vector_type(4)))  unsigned u32x4;
typedef __attribute__((ext_vector_type(8)))  unsigned u32x8;

union F16x16 { v16h v; v8h h[2]; };

// Load one 16x32 (A-style) or 32x16 (B-style) f16 WMMA fragment.
// rowptr points at this lane's row/col at k-offset 0 of the 32-chunk,
// with K contiguous. Lanes 0-15 take K {0..7,16..23}, lanes 16-31 take
// K {8..15,24..31} per the CDNA5 16-bit operand layout.
static __device__ inline v16h load_frag(const _Float16* rowptr, int lane) {
  const int off = (lane & 16) ? 8 : 0;
  F16x16 f;
  f.h[0] = *(const v8h*)(rowptr + off);
  f.h[1] = *(const v8h*)(rowptr + off + 16);
  return f.v;
}

static __device__ inline v8f wmma_f16(v16h a, v16h b, v8f c) {
  return __builtin_amdgcn_wmma_f32_16x16x32_f16(false, a, false, b, (short)0, c,
                                                false, false);
}

static __device__ inline unsigned rfl(unsigned x) {
  return __builtin_amdgcn_readfirstlane(x);
}

// ---- Tensor Data Mover: one 2-D tile (f16 elements) global -> LDS ----
// D# per CDNA5 ISA 8.3/8.4: group0 = {count/type/addrs}, group1 = dims/strides.
// tile rows are written back-to-back in LDS (tile_d0*2 bytes per row), which
// matches the row-major [rows][32] f16 staging layout used below.
static __device__ inline void tdm_load_tile_f16(unsigned lds_off,
                                                const void* gaddr,
                                                unsigned tensor_d0,
                                                unsigned tensor_d1,
                                                unsigned tile_d0,
                                                unsigned tile_d1,
                                                unsigned stride0_elems) {
  unsigned long long ga = (unsigned long long)(size_t)gaddr;
  u32x4 g0;
  g0[0] = rfl(1u);                                   // count=1, user mode
  g0[1] = rfl(lds_off);                              // lds_addr (bytes)
  g0[2] = rfl((unsigned)ga);                         // global_addr[31:0]
  g0[3] = rfl(((unsigned)(ga >> 32) & 0x01FFFFFFu) | // global_addr[56:32]
              0x80000000u);                          // type=2 ("image")
  u32x8 g1;
  g1[0] = rfl(1u << 16);                             // data_size=1 -> 2 bytes
  g1[1] = rfl((tensor_d0 & 0xFFFFu) << 16);          // tensor_dim0[15:0]
  g1[2] = rfl((tensor_d0 >> 16) | ((tensor_d1 & 0xFFFFu) << 16));
  g1[3] = rfl((tensor_d1 >> 16) | (tile_d0 << 16));  // tile_dim0
  g1[4] = rfl(tile_d1 & 0xFFFFu);                    // tile_dim1 (tile_dim2=0)
  g1[5] = rfl(stride0_elems);                        // tensor_dim0_stride[31:0]
  g1[6] = rfl(0u);                                   // stride0 hi / stride1 lo
  g1[7] = rfl(0u);                                   // stride1 hi
  asm volatile("tensor_load_to_lds %0, %1" :: "s"(g0), "s"(g1) : "memory");
}

// ---- per-lane async global -> LDS 16B copy (ASYNCcnt path) ----
static __device__ inline void async_load_b128(unsigned lds_off, const void* g) {
  asm volatile("global_load_async_to_lds_b128 %0, %1, off"
               :: "v"(lds_off), "v"(g) : "memory");
}

// ---------------- elementwise convert kernels ----------------

__global__ __launch_bounds__(256) void convert_f16(const float* __restrict__ in,
                                                   _Float16* __restrict__ out,
                                                   int n) {
  int i = blockIdx.x * 256 + threadIdx.x;
  if (i < n) out[i] = (_Float16)in[i];
}

// in: (K rows x N cols) fp32 row-major  ->  out: (N x K) f16 row-major
__global__ __launch_bounds__(256) void convert_transpose(const float* __restrict__ in,
                                                         _Float16* __restrict__ out,
                                                         int K, int N) {
  int idx = blockIdx.x * 256 + threadIdx.x;
  if (idx >= K * N) return;
  int k = idx / N, n = idx - k * N;
  out[(size_t)n * K + k] = (_Float16)in[idx];
}

// v^T: vT[b][kv][d][t] = kv_proj[b][t][slot=1][kv][d], t padded to TP with zeros
__global__ __launch_bounds__(256) void vtranspose(const float* __restrict__ kvf,
                                                  _Float16* __restrict__ vT) {
  size_t idx = (size_t)blockIdx.x * 256 + threadIdx.x;
  int t  = (int)(idx % TP);
  int d  = (int)((idx / TP) % HD);
  int kv = (int)((idx / ((size_t)TP * HD)) % NKV);
  int b  = (int)(idx / ((size_t)TP * HD * NKV));
  _Float16 val = (_Float16)0.f;
  if (t < TT)
    val = (_Float16)kvf[(size_t)(b * TT + t) * (2 * NKV * HD) + (NKV + kv) * HD + d];
  vT[idx] = val;
}

// ---------------- RMS norm + RoPE (+ q scale) ----------------
// One 128-thread block per (b, t, head). heads = NH for q, NKV for k.
__global__ __launch_bounds__(128) void normrope(const float* __restrict__ src,
                                                const float* __restrict__ scale,
                                                const int* __restrict__ positions,
                                                _Float16* __restrict__ out,
                                                int heads, float qmul) {
  __shared__ float red[128];
  int g = blockIdx.x;
  int h = g % heads;
  int t = (g / heads) % TT;
  int b = g / (heads * TT);
  int i = threadIdx.x;

  const float* row = src + (size_t)(b * TT + t) * 2048 + h * HD;
  float x1 = row[i];
  float x2 = row[i + 128];
  red[i] = x1 * x1 + x2 * x2;
  __syncthreads();
  #pragma unroll
  for (int s = 64; s > 0; s >>= 1) {
    if (i < s) red[i] += red[i + s];
    __syncthreads();
  }
  float inv = rsqrtf(red[0] * (1.0f / HD) + 1e-6f);
  float n1 = x1 * inv * (1.f + scale[i]);
  float n2 = x2 * inv * (1.f + scale[i + 128]);

  float pos = (float)positions[b * TT + t];
  float ts  = __powf(10000.f, (2.f * i) / (float)HD);
  float rad = pos / ts;
  float sn, cs;
  __sincosf(rad, &sn, &cs);
  float o1 = (n1 * cs - n2 * sn) * qmul;
  float o2 = (n2 * cs + n1 * sn) * qmul;

  _Float16* orow = out + (size_t)((b * heads + h) * TT + t) * HD;
  orow[i]       = (_Float16)o1;
  orow[i + 128] = (_Float16)o2;
}

// ---------------- tiled WMMA GEMM: C(f32) = A(M,K;f16) * Bt(N,K;f16)^T ------
// 128x128 block tile, 256 threads = 8 waves, each wave 32x64 (2x4 WMMA tiles).
// A tile staged by the Tensor Data Mover (TENSORcnt), B tile staged by
// per-lane async global->LDS copies (ASYNCcnt): no VGPR round trip.
__global__ __launch_bounds__(256) void gemm_f16(const _Float16* __restrict__ A,
                                                const _Float16* __restrict__ Bt,
                                                float* __restrict__ C,
                                                int M, int N, int K) {
  __shared__ _Float16 As[128 * 32];
  __shared__ _Float16 Bs[128 * 32];
  const int tid  = threadIdx.x;
  const int lane = tid & 31;
  const int w    = tid >> 5;
  const int wm   = w >> 1;       // 4 wave rows
  const int wn   = w & 1;        // 2 wave cols
  const int l16  = lane & 15;
  const int mtiles = M >> 7;
  const int m0 = (blockIdx.x % mtiles) << 7;
  const int n0 = (blockIdx.x / mtiles) << 7;

  const unsigned as_base = (unsigned)(size_t)(void*)&As[0];
  const unsigned bs_base = (unsigned)(size_t)(void*)&Bs[0];

  v8f acc[2][4];
  v8f z = {};
  #pragma unroll
  for (int mt = 0; mt < 2; ++mt)
    #pragma unroll
    for (int nt = 0; nt < 4; ++nt) acc[mt][nt] = z;

  for (int kb = 0; kb < K; kb += 32) {
    // A tile 128x32: single TDM descriptor, issued by wave 0 only
    if (w == 0) {
      tdm_load_tile_f16(as_base, &A[(size_t)m0 * K + kb],
                        (unsigned)K, (unsigned)M, 32u, 128u, (unsigned)K);
    }
    // B tile 128x32: 512 x 16B async copies, 2 per thread
    #pragma unroll
    for (int it = 0; it < 2; ++it) {
      int c   = it * 256 + tid;
      int row = c >> 2;
      int qd  = c & 3;
      unsigned loff = bs_base + (unsigned)(row * 32 + qd * 8) * 2u;
      async_load_b128(loff, &Bt[(size_t)(n0 + row) * K + kb + qd * 8]);
    }
    if (kb + 32 < K) {
      __builtin_prefetch(&A[(size_t)(m0 + (tid >> 1)) * K + kb + 32], 0, 0);
      __builtin_prefetch(&Bt[(size_t)(n0 + (tid >> 1)) * K + kb + 32], 0, 0);
    }
    asm volatile("s_wait_asynccnt 0" ::: "memory");
    if (w == 0) __builtin_amdgcn_s_wait_tensorcnt(0);
    __syncthreads();

    v16h af[2], bf[4];
    #pragma unroll
    for (int mt = 0; mt < 2; ++mt)
      af[mt] = load_frag(&As[(wm * 32 + mt * 16 + l16) * 32], lane);
    #pragma unroll
    for (int nt = 0; nt < 4; ++nt)
      bf[nt] = load_frag(&Bs[(wn * 64 + nt * 16 + l16) * 32], lane);
    #pragma unroll
    for (int mt = 0; mt < 2; ++mt)
      #pragma unroll
      for (int nt = 0; nt < 4; ++nt)
        acc[mt][nt] = wmma_f16(af[mt], bf[nt], acc[mt][nt]);
    __syncthreads();
  }

  const int rhalf = (lane & 16) ? 8 : 0;
  #pragma unroll
  for (int mt = 0; mt < 2; ++mt)
    #pragma unroll
    for (int nt = 0; nt < 4; ++nt)
      #pragma unroll
      for (int j = 0; j < 8; ++j) {
        int r  = m0 + wm * 32 + mt * 16 + j + rhalf;
        int cc = n0 + wn * 64 + nt * 16 + l16;
        C[(size_t)r * N + cc] = acc[mt][nt][j];
      }
}

// ---------------- sliding-window flash attention ----------------
// One wave per (b, head, 16-query-row tile). Online softmax, full 16x256
// f32 accumulator in registers (16 WMMA C-tiles).
__global__ __launch_bounds__(256) void flash_attn(const _Float16* __restrict__ qh,
                                                  const _Float16* __restrict__ kh,
                                                  const _Float16* __restrict__ vT,
                                                  _Float16* __restrict__ ench) {
  __shared__ _Float16 Pt[8][16 * 32];  // per-wave P-tile relayout scratch
  const int tid   = threadIdx.x;
  const int lane  = tid & 31;
  const int w     = tid >> 5;
  const int l16   = lane & 15;
  const int rhalf = (lane & 16) ? 8 : 0;
  const int QT    = TT / 16;

  int wg  = blockIdx.x * 8 + w;
  int qt  = wg % QT;
  int h   = (wg / QT) % NH;
  int b   = wg / (QT * NH);
  int kvh = h / GRP;
  int t0  = qt * 16;

  const _Float16* qbase = qh + (size_t)((b * NH + h) * TT) * HD;
  const _Float16* kbase = kh + (size_t)((b * NKV + kvh) * TT) * HD;
  const _Float16* vbase = vT + (size_t)((b * NKV + kvh) * HD) * TP;

  // Q A-fragments for the whole 16x256 row block (8 K-chunks of 32)
  v16h qf[8];
  #pragma unroll
  for (int kk = 0; kk < 8; ++kk)
    qf[kk] = load_frag(qbase + (size_t)(t0 + l16) * HD + kk * 32, lane);

  v8f acc[16];
  v8f z = {};
  #pragma unroll
  for (int nt = 0; nt < 16; ++nt) acc[nt] = z;
  float m[8], lsum[8];
  #pragma unroll
  for (int j = 0; j < 8; ++j) { m[j] = -1e38f; lsum[j] = 0.f; }

  int slo = t0 - (WIN - 1);
  if (slo < 0) slo = 0;
  slo &= ~31;

  for (int sb = slo; sb <= t0 + 15; sb += 32) {
    float val[2][8];
    #pragma unroll
    for (int sub = 0; sub < 2; ++sub) {
      int s0 = sb + sub * 16;
      int sc = s0 + l16;
      if (sc > TT - 1) sc = TT - 1;   // clamp load; mask handles semantics
      v8f lg = z;
      #pragma unroll
      for (int kk = 0; kk < 8; ++kk) {
        v16h bf = load_frag(kbase + (size_t)sc * HD + kk * 32, lane);
        lg = wmma_f16(qf[kk], bf, lg);
      }
      int scol = s0 + l16;
      #pragma unroll
      for (int j = 0; j < 8; ++j) {
        int tr = t0 + j + rhalf;
        bool ok = (scol <= tr) && ((tr - scol) < WIN);
        val[sub][j] = ok ? lg[j] : BIG_NEG;
      }
    }
    // per-row chunk max (16-lane halves match the C-layout row split)
    #pragma unroll
    for (int j = 0; j < 8; ++j) {
      float mv = fmaxf(val[0][j], val[1][j]);
      mv = fmaxf(mv, __shfl_xor(mv, 8, 32));
      mv = fmaxf(mv, __shfl_xor(mv, 4, 32));
      mv = fmaxf(mv, __shfl_xor(mv, 2, 32));
      mv = fmaxf(mv, __shfl_xor(mv, 1, 32));
      float mn  = fmaxf(m[j], mv);
      float scl = __expf(m[j] - mn);
      m[j] = mn;
      lsum[j] *= scl;
      #pragma unroll
      for (int nt = 0; nt < 16; ++nt) acc[nt][j] *= scl;
    }
    // probabilities -> LDS (C-layout -> A-layout relayout) + row sums
    #pragma unroll
    for (int sub = 0; sub < 2; ++sub) {
      #pragma unroll
      for (int j = 0; j < 8; ++j) {
        float p = __expf(val[sub][j] - m[j]);   // underflows to 0 when masked
        Pt[w][(j + rhalf) * 32 + sub * 16 + l16] = (_Float16)p;
        float r = p;
        r += __shfl_xor(r, 8, 32);
        r += __shfl_xor(r, 4, 32);
        r += __shfl_xor(r, 2, 32);
        r += __shfl_xor(r, 1, 32);
        lsum[j] += r;
      }
    }
    v16h pf = load_frag(&Pt[w][l16 * 32], lane);
    #pragma unroll
    for (int nt = 0; nt < 16; ++nt) {
      v16h vf = load_frag(vbase + (size_t)(nt * 16 + l16) * TP + sb, lane);
      acc[nt] = wmma_f16(pf, vf, acc[nt]);
    }
  }

  // normalize and emit encoded (b, t, h*HD+d) as f16 for the final GEMM
  #pragma unroll
  for (int j = 0; j < 8; ++j) {
    float inv = 1.0f / lsum[j];
    int tr = t0 + j + rhalf;
    _Float16* orow = ench + (size_t)(b * TT + tr) * WID + h * HD;
    #pragma unroll
    for (int nt = 0; nt < 16; ++nt)
      orow[nt * 16 + l16] = (_Float16)(acc[nt][j] * inv);
  }
}

// ---------------- host-side launcher ----------------

extern "C" void kernel_launch(void* const* d_in, const int* in_sizes, int n_in,
                              void* d_out, int out_size, void* d_ws, size_t ws_size,
                              hipStream_t stream) {
  (void)in_sizes; (void)n_in; (void)out_size; (void)ws_size;
  const float* x    = (const float*)d_in[0];
  const int* pos    = (const int*)d_in[1];
  // d_in[2] = attn_mask: reconstructed analytically (causal & |dt|<WINDOW)
  const float* wq   = (const float*)d_in[3];
  const float* wkv  = (const float*)d_in[4];
  const float* wout = (const float*)d_in[5];
  const float* qsc  = (const float*)d_in[6];
  const float* ksc  = (const float*)d_in[7];
  float* out = (float*)d_out;

  const int BT = BB * TT;  // 4096

  // workspace carve-up
  _Float16* xh    = (_Float16*)d_ws;
  _Float16* wqT   = xh    + (size_t)BT * WID;
  _Float16* wkvT  = wqT   + (size_t)WID * WID;
  _Float16* woutT = wkvT  + (size_t)WID * WID;
  float*    qf    = (float*)(woutT + (size_t)WID * WID);
  float*    kvf   = qf    + (size_t)BT * WID;
  _Float16* qhp   = (_Float16*)(kvf + (size_t)BT * WID);
  _Float16* khp   = qhp   + (size_t)BB * NH * TT * HD;
  _Float16* vTp   = khp   + (size_t)BB * NKV * TT * HD;
  _Float16* ench  = vTp   + (size_t)BB * NKV * HD * TP;

  // 1) precision conversion / weight transposition
  convert_f16<<<(BT * WID) / 256, 256, 0, stream>>>(x, xh, BT * WID);
  convert_transpose<<<(WID * WID) / 256, 256, 0, stream>>>(wq, wqT, WID, WID);
  convert_transpose<<<(WID * WID) / 256, 256, 0, stream>>>(wkv, wkvT, WID, WID);
  convert_transpose<<<(WID * WID) / 256, 256, 0, stream>>>(wout, woutT, WID, WID);

  // 2) q / kv projections (WMMA GEMMs, TDM + async-LDS staging)
  int gemm_grid = (BT / 128) * (WID / 128);
  gemm_f16<<<gemm_grid, 256, 0, stream>>>(xh, wqT, qf, BT, WID, WID);
  gemm_f16<<<gemm_grid, 256, 0, stream>>>(xh, wkvT, kvf, BT, WID, WID);

  // 3) RMS-norm + RoPE (+ 1/sqrt(D) folded into q); v transposed to (d, t)
  normrope<<<BT * NH, 128, 0, stream>>>(qf, qsc, pos, qhp, NH, 0.0625f);
  normrope<<<BT * NKV, 128, 0, stream>>>(kvf, ksc, pos, khp, NKV, 1.0f);
  vtranspose<<<(BB * NKV * HD * TP) / 256, 256, 0, stream>>>(kvf, vTp);

  // 4) sliding-window flash attention (WMMA QK^T + PV)
  flash_attn<<<(BB * NH * (TT / 16)) / 8, 256, 0, stream>>>(qhp, khp, vTp, ench);

  // 5) output projection straight into d_out
  gemm_f16<<<gemm_grid, 256, 0, stream>>>(ench, woutT, out, BT, WID, WID);
}
